// MultiHeadSelfAttention_3358664425917
// MI455X (gfx1250) — matrix operand order repaired and checked by reading, not compiled, not run on hardware
//
#include <hip/hip_runtime.h>
#include <hip/hip_bf16.h>
#include <cstdint>

#define D_MODEL 1024
#define NUM_HEADS 16
#define DEPTH 64
#define BATCH 2
#define SEQ 2048
#define TOKENS (BATCH * SEQ) // 4096

typedef __attribute__((ext_vector_type(16))) __bf16 v16bf;
typedef __attribute__((ext_vector_type(8)))  float  v8f;

#if __has_builtin(__builtin_amdgcn_tensor_load_to_lds)
#define HAVE_TDM 1
typedef __attribute__((ext_vector_type(4))) unsigned int v4u;
typedef __attribute__((ext_vector_type(8))) int v8i;
typedef __attribute__((ext_vector_type(4))) int v4i;
#else
#define HAVE_TDM 0
#endif

union Frag16 { unsigned int u[8]; v16bf v; };

__device__ __forceinline__ unsigned short f2bf(float f) {
  unsigned int x = __float_as_uint(f);
  x += 0x7fffu + ((x >> 16) & 1u); // round-to-nearest-even
  return (unsigned short)(x >> 16);
}
__device__ __forceinline__ unsigned int pack2bf(float lo, float hi) {
  return (unsigned int)f2bf(lo) | ((unsigned int)f2bf(hi) << 16);
}

// 16-bit A-matrix 16x32 (MxK) VGPR layout: v0..3 -> K = g*8 + 2v, v4..7 -> K = 16 + g*8 + 2(v-4)
__device__ __forceinline__ int kA(int v, int g) {
  return (v < 4) ? (g * 8 + 2 * v) : (16 + g * 8 + 2 * (v - 4));
}
// 16-bit B-matrix 32x16 (KxN) VGPR layout: v0..3 -> K = g*8 + 2v, v4..7 -> K = 16 + g*8 + 2(v-4)
__device__ __forceinline__ int kB(int v, int g) {
  return (v < 4) ? (g * 8 + 2 * v) : (16 + g * 8 + 2 * (v - 4));
}

// ---------------------------------------------------------------- conversions
__global__ void f32_to_bf16_kernel(const float* __restrict__ src,
                                   unsigned short* __restrict__ dst, int n) {
  int i = blockIdx.x * blockDim.x + threadIdx.x;
  if (i < n) dst[i] = f2bf(src[i]);
}

// V [token][1024] (head cols h*64+d) -> Vt [b][h][d][kk]  (contiguous in kk)
__global__ void vtranspose_kernel(const unsigned short* __restrict__ V,
                                  unsigned short* __restrict__ Vt) {
  int idx = blockIdx.x * blockDim.x + threadIdx.x; // over B*H*DEPTH*SEQ = 4,194,304
  int kk = idx & (SEQ - 1);
  int t = idx >> 11;
  int d = t & (DEPTH - 1);  t >>= 6;
  int h = t & (NUM_HEADS - 1);
  int b = t >> 4;
  Vt[idx] = V[(size_t)(b * SEQ + kk) * D_MODEL + h * DEPTH + d];
}

// ---------------------------------------------------------------- bf16 GEMM
// D[M,N] = A[M,K] * B[K,N] + bias; A,B bf16 row-major; out f32 (Df) or bf16 (Dbf).
// M % 128 == 0, N % 64 == 0, K % 32 == 0 (true for all calls here).
#define GBM 128
#define GBN 64
#define GBK 32
#define LDA 34
#define LDB 34

__global__ void __launch_bounds__(256)
gemm_bf16_kernel(const unsigned short* __restrict__ A,
                 const unsigned short* __restrict__ B,
                 const float* __restrict__ bias,
                 float* __restrict__ Df,
                 unsigned short* __restrict__ Dbf,
                 int M, int N, int K) {
  __shared__ unsigned short As[GBM * LDA];
  __shared__ unsigned short Bs[GBN * LDB];

  const int tid  = threadIdx.x;
  const int lane = tid & 31;
  const int w    = tid >> 5;  // 0..7
  const int wm   = w & 3;     // wave M tile (0..3)
  const int wn   = w >> 2;    // wave N tile (0..1)
  const int n15  = lane & 15;
  const int g    = lane >> 4;

  const int m0 = blockIdx.y * GBM;
  const int n0 = blockIdx.x * GBN;

  v8f acc[2][2];
  for (int mt = 0; mt < 2; ++mt)
    for (int nt = 0; nt < 2; ++nt)
      acc[mt][nt] = (v8f){0.f, 0.f, 0.f, 0.f, 0.f, 0.f, 0.f, 0.f};

  const int bkk  = tid >> 3;        // 0..31
  const int bns  = (tid & 7) * 8;   // 0..56
#if !HAVE_TDM
  const int arow = tid >> 1;        // 0..127
  const int akh  = (tid & 1) * 16;  // 0 or 16
#else
  // LDS byte offset of As for the TDM descriptor (generic -> LDS addrspace cast)
  const unsigned lds_as_base =
      (unsigned)(uintptr_t)(__attribute__((address_space(3))) unsigned short*)As;
#endif

  for (int k0 = 0; k0 < K; k0 += GBK) {
#if HAVE_TDM
    // ---- Tensor Data Mover: DMA the 128x32 bf16 A tile into LDS.
    // D# per CDNA5 ISA: 2D tile, data_size=2B,
    // tile_dim0=32 (K), tile_dim1=128 (M rows), row = 64B = 8 qwords ->
    // pad_interval=3, pad_amount=0 adds 1 dword per row => LDS row stride 34
    // elements (= LDA), matching the bank-conflict-free fragment layout.
    if (w == 0) {
      unsigned long long ga =
          (unsigned long long)(uintptr_t)(A + (size_t)m0 * K + k0); // tile byte addr
      v4u g0;
      g0.x = 1u;                                            // count=1, user mode
      g0.y = lds_as_base;                                   // lds_addr (bytes)
      g0.z = (unsigned)(ga & 0xffffffffu);                  // global_addr[31:0]
      g0.w = (unsigned)((ga >> 32) & 0x01ffffffu) | (2u << 30); // addr[56:32] | type=2
      v8i g1;
      g1[0] = (int)((1u << 16) | (1u << 20) | (3u << 22));  // data_size=2B, pad_en, intv=8qw
      g1[1] = (int)(((unsigned)K & 0xffffu) << 16);         // tensor_dim0[15:0]
      g1[2] = (int)((((unsigned)K >> 16) & 0xffffu) |
                    (((unsigned)M & 0xffffu) << 16));       // dim0[31:16] | dim1[15:0]
      g1[3] = (int)((((unsigned)M >> 16) & 0xffffu) |
                    (32u << 16));                           // dim1[31:16] | tile_dim0=32
      g1[4] = (int)128;                                     // tile_dim1=128, tile_dim2=0
      g1[5] = (int)K;                                       // tensor_dim0_stride[31:0]
      g1[6] = 0;                                            // stride hi | dim1_stride lo
      g1[7] = 0;
      v4i gz4 = (v4i){0, 0, 0, 0};
      v8i gz8 = (v8i){0, 0, 0, 0, 0, 0, 0, 0};
      // 6-arg form (this toolchain): (g0, g1, g2, g3, g4, cpol)
      __builtin_amdgcn_tensor_load_to_lds(g0, g1, gz4, gz4, gz8, 0);
#if __has_builtin(__builtin_amdgcn_s_wait_tensorcnt)
      __builtin_amdgcn_s_wait_tensorcnt(0);
#else
      asm volatile("s_wait_tensorcnt 0x0" ::: "memory");
#endif
    }
#else
    // ---- fallback: stage A tile through VGPRs (row-major, 128x32)
    const uint4* ag = (const uint4*)(A + (size_t)(m0 + arow) * K + k0 + akh);
    uint4 q0 = ag[0], q1 = ag[1];
    unsigned int* asw = (unsigned int*)&As[arow * LDA + akh];
    asw[0] = q0.x; asw[1] = q0.y; asw[2] = q0.z; asw[3] = q0.w;
    asw[4] = q1.x; asw[5] = q1.y; asw[6] = q1.z; asw[7] = q1.w;
#endif
    // stage B tile transposed into LDS as [n][k] so fragment reads are u32 K-pairs
    uint4 qb = *(const uint4*)(B + (size_t)(k0 + bkk) * N + n0 + bns);
    unsigned int bw[4] = {qb.x, qb.y, qb.z, qb.w};
#pragma unroll
    for (int j = 0; j < 4; ++j) {
      Bs[(bns + 2 * j + 0) * LDB + bkk] = (unsigned short)(bw[j] & 0xffffu);
      Bs[(bns + 2 * j + 1) * LDB + bkk] = (unsigned short)(bw[j] >> 16);
    }
    if (k0 + GBK < K)
      __builtin_prefetch(B + (size_t)(k0 + GBK + bkk) * N + n0 + bns, 0, 3);
    __syncthreads();

    Frag16 af[2], bf[2];
#pragma unroll
    for (int mt = 0; mt < 2; ++mt) {
      int row = wm * 32 + mt * 16 + n15;
#pragma unroll
      for (int v = 0; v < 8; ++v)
        af[mt].u[v] = *(const unsigned int*)&As[row * LDA + kA(v, g)];
    }
#pragma unroll
    for (int nt = 0; nt < 2; ++nt) {
      int col = wn * 32 + nt * 16 + n15;
#pragma unroll
      for (int v = 0; v < 8; ++v)
        bf[nt].u[v] = *(const unsigned int*)&Bs[col * LDB + kB(v, g)];
    }
#pragma unroll
    for (int mt = 0; mt < 2; ++mt)
#pragma unroll
      for (int nt = 0; nt < 2; ++nt)
        acc[mt][nt] = __builtin_amdgcn_wmma_f32_16x16x32_bf16(
            false, af[mt].v, false, bf[nt].v, (short)0, acc[mt][nt], false, false);
    __syncthreads();
  }

  // epilogue: C/D layout — lane(0-15)=N, vgpr r: M=r (g=0) / r+8 (g=1)
  for (int mt = 0; mt < 2; ++mt) {
    for (int nt = 0; nt < 2; ++nt) {
      int col = n0 + wn * 32 + nt * 16 + n15;
      float bvv = bias[col];
#pragma unroll
      for (int r = 0; r < 8; ++r) {
        int row = m0 + wm * 32 + mt * 16 + g * 8 + r;
        float val = acc[mt][nt][r] + bvv;
        if (Dbf) Dbf[(size_t)row * N + col] = f2bf(val);
        else     Df [(size_t)row * N + col] = val;
      }
    }
  }
}

// ---------------------------------------------------------------- attention
// One workgroup per (b, h, 32 query rows). Scores live in 256 KB dynamic LDS
// (CDNA5: up to 320 KB per workgroup). 8 waves, wave32.
#define SC_STRIDE 2052  // 32 f32 rows, padded: lane stride 2052 % 64 banks = 4 -> conflict-free

__global__ void __launch_bounds__(256)
attn_kernel(const unsigned short* __restrict__ Qb,
            const unsigned short* __restrict__ Kb,
            const unsigned short* __restrict__ Vt,
            float* __restrict__ attn,
            unsigned short* __restrict__ ctx) {
  extern __shared__ float sc[]; // [32][SC_STRIDE]
  const int tid  = threadIdx.x;
  const int lane = tid & 31;
  const int w    = tid >> 5;
  const int n15  = lane & 15;
  const int g    = lane >> 4;

  const int qt = blockIdx.x;     // 0..63
  const int h  = blockIdx.y;     // 0..15
  const int b  = blockIdx.z;     // 0..1
  const int q0 = qt * 32;

  // ---- Phase 1: logits = scale * Q K^T, scores -> LDS ----
  Frag16 aq[2][2];
#pragma unroll
  for (int mt = 0; mt < 2; ++mt) {
    const unsigned short* qrow =
        Qb + (size_t)(b * SEQ + q0 + mt * 16 + n15) * D_MODEL + h * DEPTH;
#pragma unroll
    for (int kc = 0; kc < 2; ++kc)
#pragma unroll
      for (int v = 0; v < 8; ++v)
        aq[mt][kc].u[v] = *(const unsigned int*)(qrow + kc * 32 + kA(v, g));
  }

  const float scale = 0.125f; // 1/sqrt(64)
  for (int nt = w; nt < SEQ / 16; nt += 8) {
    Frag16 bk[2];
    // B = K^T: lane column = key index, K-dim = d (contiguous in memory)
    const unsigned short* krow =
        Kb + (size_t)(b * SEQ + nt * 16 + n15) * D_MODEL + h * DEPTH;
#pragma unroll
    for (int kc = 0; kc < 2; ++kc)
#pragma unroll
      for (int v = 0; v < 8; ++v)
        bk[kc].u[v] = *(const unsigned int*)(krow + kc * 32 + kB(v, g));
#pragma unroll
    for (int mt = 0; mt < 2; ++mt) {
      v8f acc = (v8f){0.f, 0.f, 0.f, 0.f, 0.f, 0.f, 0.f, 0.f};
#pragma unroll
      for (int kc = 0; kc < 2; ++kc)
        acc = __builtin_amdgcn_wmma_f32_16x16x32_bf16(
            false, aq[mt][kc].v, false, bk[kc].v, (short)0, acc, false, false);
#pragma unroll
      for (int r = 0; r < 8; ++r)
        sc[(mt * 16 + g * 8 + r) * SC_STRIDE + nt * 16 + n15] = acc[r] * scale;
    }
  }
  __syncthreads();

  // ---- Phase 2: fp32 softmax per row (8 lanes / row), write attn to d_out ----
  {
    const int row = tid >> 3;
    const int sub = tid & 7;
    float* srow = sc + row * SC_STRIDE + sub * 256;
    float mx = -3.4e38f;
    for (int i = 0; i < 256; ++i) mx = fmaxf(mx, srow[i]);
    for (int m = 1; m < 8; m <<= 1) mx = fmaxf(mx, __shfl_xor(mx, m, 32));
    float sum = 0.f;
    for (int i = 0; i < 256; ++i) {
      float e = __expf(srow[i] - mx);
      srow[i] = e;
      sum += e;
    }
    for (int m = 1; m < 8; m <<= 1) sum += __shfl_xor(sum, m, 32);
    const float inv = 1.f / sum;
    float* gout =
        attn + ((size_t)((b * NUM_HEADS + h) * SEQ) + q0 + row) * SEQ + sub * 256;
    for (int i = 0; i < 256; ++i) {
      float a = srow[i] * inv;
      srow[i] = a;
      gout[i] = a;
    }
  }
  __syncthreads();

  // ---- Phase 3: ctx = attn * V. A from LDS (f32 -> bf16 pack), B from Vt ----
  {
    const int mt = w >> 2;               // 0..1 (query sub-tile)
    const int nt = w & 3;                // 0..3 (d sub-tile)
    const int m  = mt * 16 + n15;        // A row (query)
    const int d  = nt * 16 + n15;        // B column (depth)
    const unsigned short* vrow =
        Vt + (size_t)((b * NUM_HEADS + h) * DEPTH + d) * SEQ;
    v8f acc = (v8f){0.f, 0.f, 0.f, 0.f, 0.f, 0.f, 0.f, 0.f};
    for (int k0 = 0; k0 < SEQ; k0 += 32) {
      Frag16 a, bf;
      const float* srow = sc + m * SC_STRIDE + k0;
#pragma unroll
      for (int v = 0; v < 8; ++v) {
        int k = kA(v, g);
        a.u[v] = pack2bf(srow[k], srow[k + 1]);
      }
#pragma unroll
      for (int v = 0; v < 8; ++v)
        bf.u[v] = *(const unsigned int*)(vrow + k0 + kB(v, g));
      acc = __builtin_amdgcn_wmma_f32_16x16x32_bf16(
          false, a.v, false, bf.v, (short)0, acc, false, false);
    }
#pragma unroll
    for (int r = 0; r < 8; ++r) {
      int token = b * SEQ + q0 + mt * 16 + g * 8 + r;
      ctx[(size_t)token * D_MODEL + h * DEPTH + nt * 16 + n15] = f2bf(acc[r]);
    }
  }
}

// ---------------------------------------------------------------- launch
extern "C" void kernel_launch(void* const* d_in, const int* in_sizes, int n_in,
                              void* d_out, int out_size, void* d_ws, size_t ws_size,
                              hipStream_t stream) {
  (void)in_sizes; (void)n_in; (void)out_size; (void)ws_size;
  const float* x  = (const float*)d_in[0];
  const float* wq = (const float*)d_in[1];
  const float* bq = (const float*)d_in[2];
  const float* wk = (const float*)d_in[3];
  const float* bk = (const float*)d_in[4];
  const float* wv = (const float*)d_in[5];
  const float* bv = (const float*)d_in[6];
  const float* wo = (const float*)d_in[7];
  const float* bo = (const float*)d_in[8];

  float* out  = (float*)d_out;                    // [2,2048,1024]
  float* attn = out + (size_t)TOKENS * D_MODEL;   // [2,16,2048,2048]

  char* ws = (char*)d_ws;
  const size_t MB = 1024 * 1024;
  unsigned short* Xb  = (unsigned short*)(ws + 0 * MB);   // 8 MB
  unsigned short* Wqb = (unsigned short*)(ws + 8 * MB);   // 2 MB
  unsigned short* Wkb = (unsigned short*)(ws + 10 * MB);  // 2 MB
  unsigned short* Wvb = (unsigned short*)(ws + 12 * MB);  // 2 MB
  unsigned short* Wob = (unsigned short*)(ws + 14 * MB);  // 2 MB
  unsigned short* Qb  = (unsigned short*)(ws + 16 * MB);  // 8 MB
  unsigned short* Kbm = (unsigned short*)(ws + 24 * MB);  // 8 MB
  unsigned short* Vb  = (unsigned short*)(ws + 32 * MB);  // 8 MB
  unsigned short* Vt  = (unsigned short*)(ws + 40 * MB);  // 8 MB
  unsigned short* Ctx = (unsigned short*)(ws + 48 * MB);  // 8 MB  (56 MB total)

  const int nx = TOKENS * D_MODEL;
  const int nw = D_MODEL * D_MODEL;
  f32_to_bf16_kernel<<<(nx + 255) / 256, 256, 0, stream>>>(x, Xb, nx);
  f32_to_bf16_kernel<<<(nw + 255) / 256, 256, 0, stream>>>(wq, Wqb, nw);
  f32_to_bf16_kernel<<<(nw + 255) / 256, 256, 0, stream>>>(wk, Wkb, nw);
  f32_to_bf16_kernel<<<(nw + 255) / 256, 256, 0, stream>>>(wv, Wvb, nw);
  f32_to_bf16_kernel<<<(nw + 255) / 256, 256, 0, stream>>>(wo, Wob, nw);

  dim3 ggrid(D_MODEL / GBN, TOKENS / GBM); // (16, 32)
  gemm_bf16_kernel<<<ggrid, 256, 0, stream>>>(Xb, Wqb, bq, nullptr, Qb,  TOKENS, D_MODEL, D_MODEL);
  gemm_bf16_kernel<<<ggrid, 256, 0, stream>>>(Xb, Wkb, bk, nullptr, Kbm, TOKENS, D_MODEL, D_MODEL);
  gemm_bf16_kernel<<<ggrid, 256, 0, stream>>>(Xb, Wvb, bv, nullptr, Vb,  TOKENS, D_MODEL, D_MODEL);

  vtranspose_kernel<<<(TOKENS * D_MODEL) / 256, 256, 0, stream>>>(Vb, Vt);

  const int smem = 32 * SC_STRIDE * 4; // 262,656 B dynamic LDS (<= 320 KB/WGP)
  (void)hipFuncSetAttribute((const void*)attn_kernel,
                            hipFuncAttributeMaxDynamicSharedMemorySize, smem);
  attn_kernel<<<dim3(SEQ / 32, NUM_HEADS, BATCH), 256, smem, stream>>>(
      Qb, Kbm, Vt, attn, Ctx);

  gemm_bf16_kernel<<<ggrid, 256, 0, stream>>>(Ctx, Wob, bo, out, nullptr, TOKENS, D_MODEL, D_MODEL);
}
